// Model_80994493268339
// MI455X (gfx1250) — matrix operand order repaired and checked by reading, not compiled, not run on hardware
//
#include <hip/hip_runtime.h>
#include <hip/hip_bf16.h>
#include <math.h>

// Problem dims
#define TT 256
#define BB 32
#define VV 4712
#define HH 500
#define GG 2000        // 4*H (gate dim), interleaved as [i0 f0 g0 o0 i1 f1 g1 o1 ...]
#define GP 2048        // gate dim padded to 2048 (zero rows) so N splits into 256-col blocks
#define KPAD_V 4736    // V padded to multiple of 32 (zero pad -> exact GEMM)
#define KPAD_H 512     // H padded to multiple of 32
#define NROWS (TT*BB)  // 8192
#define NTILES_G 125   // 2000 / 16 (real gate tiles used by the recurrence)
#define STEPS 255

typedef __attribute__((ext_vector_type(16))) __bf16 v16bf;
typedef __attribute__((ext_vector_type(8)))  float  v8f;

union BFV { uint4 u[2]; v16bf v; };

__device__ inline unsigned short f2bf(float x) {        // fp32 -> bf16 RNE
  unsigned int u = __float_as_uint(x);
  u += 0x7FFFu + ((u >> 16) & 1u);
  return (unsigned short)(u >> 16);
}

// A tile (16x32 bf16, M x K), row-major source, per CDNA5 ISA layout:
// lane m=L&15, kh=L>>4; halves i=0..7 -> K=kh*8+i ; i=8..15 -> K=16+kh*8+(i-8)
__device__ inline v16bf load_a_tile(const unsigned short* __restrict__ A,
                                    int row0, int k0, int lane, int lda) {
  int m = lane & 15, kh = lane >> 4;
  const unsigned short* p = A + (size_t)(row0 + m) * lda + k0 + kh * 8;
  BFV t;
  t.u[0] = *(const uint4*)(p);        // K = kh*8 + 0..7
  t.u[1] = *(const uint4*)(p + 16);   // K = 16 + kh*8 + 0..7
  return t.v;
}

// B tile (32x16 bf16, K x N) where B = W^T and W is row-major (N rows, K cols):
// lane n=L&15, kg=L>>4; halves i=0..7 -> K=kg*8+i ; i=8..15 -> K=16+kg*8+(i-8)
__device__ inline v16bf load_b_tile(const unsigned short* __restrict__ W,
                                    int col0, int k0, int lane, int ldw) {
  int n = lane & 15, kg = lane >> 4;
  const unsigned short* p = W + (size_t)(col0 + n) * ldw + k0 + kg * 8;
  BFV t;
  t.u[0] = *(const uint4*)(p);
  t.u[1] = *(const uint4*)(p + 16);
  return t.v;
}

__device__ inline float sigm(float x) { return 1.0f / (1.0f + __expf(-x)); }

// ---------------- init / conversion kernels ----------------

__global__ void k_init(unsigned short* hbuf, unsigned int* bar, float* out) {
  int i = blockIdx.x * blockDim.x + threadIdx.x;
  if (i < 2 * BB * KPAD_H) hbuf[i] = 0;      // h ping-pong buffers (incl. K pad)
  if (i < 1024) bar[i] = 0;                  // per-step barrier counters
  if (i == 0) out[0] = 0.0f;                 // loss accumulator
}

__global__ void k_conv_seq(const float* __restrict__ src, unsigned short* __restrict__ dst) {
  int row = blockIdx.x;                      // 0..8191  (t*B + b)
  for (int c = threadIdx.x; c < KPAD_V; c += blockDim.x) {
    float v = (c < VV) ? src[(size_t)row * VV + c] : 0.0f;
    dst[(size_t)row * KPAD_V + c] = f2bf(v);
  }
}

// permute gate rows to interleaved order + pad K and N; dst row p=4j+g <- src row g*H+j
__global__ void k_conv_wih(const float* __restrict__ src, unsigned short* __restrict__ dst) {
  int p = blockIdx.x;                        // 0..2047 (rows >= 2000 are zero pad)
  int j = p >> 2, g = p & 3;
  const float* srow = (p < GG) ? (src + (size_t)(g * HH + j) * VV) : (const float*)0;
  for (int c = threadIdx.x; c < KPAD_V; c += blockDim.x) {
    float v = (p < GG && c < VV) ? srow[c] : 0.0f;
    dst[(size_t)p * KPAD_V + c] = f2bf(v);
  }
}

__global__ void k_conv_whh(const float* __restrict__ src, unsigned short* __restrict__ dst) {
  int p = blockIdx.x;                        // 0..1999
  int j = p >> 2, g = p & 3;
  const float* srow = src + (size_t)(g * HH + j) * HH;
  for (int c = threadIdx.x; c < KPAD_H; c += blockDim.x) {
    float v = (c < HH) ? srow[c] : 0.0f;
    dst[(size_t)p * KPAD_H + c] = f2bf(v);
  }
}

__global__ void k_bias(const float* __restrict__ bih, const float* __restrict__ bhh,
                       float* __restrict__ dst) {
  int p = blockIdx.x * blockDim.x + threadIdx.x;
  if (p < GG) { int j = p >> 2, g = p & 3; dst[p] = bih[g * HH + j] + bhh[g * HH + j]; }
}

// ---------------- big GEMM: gates_x = X(8192 x 4736) @ Wihp^T -> (8192 x 2048) ----------------
// grid (8, 128), block 256 (8 waves). Wave tile: 64 rows x 32 cols (4x2 WMMA tiles).
// All 8 waves of a block share one 64-row A panel (WGP-cache reuse);
// 12 b128 loads per 8 WMMAs.
__global__ void __launch_bounds__(256) wmma_xgates(const unsigned short* __restrict__ A,
                                                   const unsigned short* __restrict__ W,
                                                   float* __restrict__ C) {
  int lane = threadIdx.x & 31;
  int wave = threadIdx.x >> 5;
  int colw = blockIdx.x * 256 + wave * 32;   // this wave's 32 output columns
  int mbase = blockIdx.y * 64;               // block's 64-row A panel
  v8f acc00 = {}, acc01 = {}, acc10 = {}, acc11 = {};
  v8f acc20 = {}, acc21 = {}, acc30 = {}, acc31 = {};
  for (int k0 = 0; k0 < KPAD_V; k0 += 32) {
    v16bf b0 = load_b_tile(W, colw,      k0, lane, KPAD_V);
    v16bf b1 = load_b_tile(W, colw + 16, k0, lane, KPAD_V);
    v16bf a0 = load_a_tile(A, mbase,      k0, lane, KPAD_V);
    v16bf a1 = load_a_tile(A, mbase + 16, k0, lane, KPAD_V);
    v16bf a2 = load_a_tile(A, mbase + 32, k0, lane, KPAD_V);
    v16bf a3 = load_a_tile(A, mbase + 48, k0, lane, KPAD_V);
    acc00 = __builtin_amdgcn_wmma_f32_16x16x32_bf16(false, a0, false, b0, (short)0, acc00, false, false);
    acc01 = __builtin_amdgcn_wmma_f32_16x16x32_bf16(false, a0, false, b1, (short)0, acc01, false, false);
    acc10 = __builtin_amdgcn_wmma_f32_16x16x32_bf16(false, a1, false, b0, (short)0, acc10, false, false);
    acc11 = __builtin_amdgcn_wmma_f32_16x16x32_bf16(false, a1, false, b1, (short)0, acc11, false, false);
    acc20 = __builtin_amdgcn_wmma_f32_16x16x32_bf16(false, a2, false, b0, (short)0, acc20, false, false);
    acc21 = __builtin_amdgcn_wmma_f32_16x16x32_bf16(false, a2, false, b1, (short)0, acc21, false, false);
    acc30 = __builtin_amdgcn_wmma_f32_16x16x32_bf16(false, a3, false, b0, (short)0, acc30, false, false);
    acc31 = __builtin_amdgcn_wmma_f32_16x16x32_bf16(false, a3, false, b1, (short)0, acc31, false, false);
  }
  int kh = lane >> 4;
  int cn0 = colw + (lane & 15);
#pragma unroll
  for (int r = 0; r < 8; ++r) {
    size_t row0 = (size_t)(mbase + r + 8 * kh);
    C[(row0     ) * GP + cn0     ] = acc00[r];
    C[(row0     ) * GP + cn0 + 16] = acc01[r];
    C[(row0 + 16) * GP + cn0     ] = acc10[r];
    C[(row0 + 16) * GP + cn0 + 16] = acc11[r];
    C[(row0 + 32) * GP + cn0     ] = acc20[r];
    C[(row0 + 32) * GP + cn0 + 16] = acc21[r];
    C[(row0 + 48) * GP + cn0     ] = acc30[r];
    C[(row0 + 48) * GP + cn0 + 16] = acc31[r];
  }
}

// ---------------- persistent recurrence: encoder then decoder, 510 steps ----------------
// grid 125 x blockDim 32 (one wave per 16-col gate tile = 4 hidden units).
__global__ void __launch_bounds__(32) lstm_recurrence(
    const unsigned short* __restrict__ encWhh, const unsigned short* __restrict__ decWhh,
    const float* __restrict__ encBias, const float* __restrict__ decBias,
    const float* __restrict__ gEnc, const float* __restrict__ gDec,
    unsigned short* __restrict__ hbuf,   // 2 * 32 * 512 bf16 (ping-pong), pre-zeroed
    float* __restrict__ decH,            // 255 * 32 * 500 f32
    unsigned int* __restrict__ bar) {
  int lane = threadIdx.x;
  int n = blockIdx.x;                    // gate tile (cols 16n..16n+15, units 4n..4n+3)
  int col = lane & 15, kh = lane >> 4;
  int base = lane & ~3;
  int unit = 4 * n + (col >> 2);
  float c[16];
#pragma unroll
  for (int i = 0; i < 16; ++i) c[i] = 0.0f;

  int gs = 0;                            // global step 0..509 (drives ping-pong & barrier)
  for (int phase = 0; phase < 2; ++phase) {
    const unsigned short* Whh = phase ? decWhh : encWhh;
    const float* gx = phase ? gDec : gEnc;
    float bval = (phase ? decBias : encBias)[16 * n + col];
    for (int s = 0; s < STEPS; ++s, ++gs) {
      int t = phase ? s : (STEPS - s);   // encoder consumes sequence[255..1]
      const unsigned short* hrd = hbuf + (size_t)(gs & 1) * BB * KPAD_H;
      unsigned short*       hwr = hbuf + (size_t)((gs + 1) & 1) * BB * KPAD_H;

      v8f acc0 = {}, acc1 = {};
      for (int k0 = 0; k0 < KPAD_H; k0 += 32) {
        v16bf b  = load_b_tile(Whh, 16 * n, k0, lane, KPAD_H);
        v16bf a0 = load_a_tile(hrd, 0,  k0, lane, KPAD_H);    // batches 0..15
        v16bf a1 = load_a_tile(hrd, 16, k0, lane, KPAD_H);    // batches 16..31
        acc0 = __builtin_amdgcn_wmma_f32_16x16x32_bf16(false, a0, false, b, (short)0, acc0, false, false);
        acc1 = __builtin_amdgcn_wmma_f32_16x16x32_bf16(false, a1, false, b, (short)0, acc1, false, false);
      }

      // gates = x-projection (precomputed) + h-projection + (bih+bhh)
      const float* gxt = gx + (size_t)t * BB * GP + 16 * n + col;
      float g[16];
#pragma unroll
      for (int r = 0; r < 8; ++r) {
        g[r]     = acc0[r] + bval + gxt[(size_t)(kh * 8 + r) * GP];
        g[8 + r] = acc1[r] + bval + gxt[(size_t)(16 + kh * 8 + r) * GP];
      }

      // gather i,f,g,o from the 4 adjacent lanes of this unit; update c, h
#pragma unroll
      for (int si = 0; si < 16; ++si) {
        float iv = __shfl(g[si], base + 0, 32);
        float fv = __shfl(g[si], base + 1, 32);
        float gv = __shfl(g[si], base + 2, 32);
        float ov = __shfl(g[si], base + 3, 32);
        float cn_ = sigm(fv) * c[si] + sigm(iv) * tanhf(gv);
        c[si] = cn_;
        float hv = sigm(ov) * tanhf(cn_);
        if ((lane & 3) == 0) {
          int batch = (si < 8) ? (kh * 8 + si) : (16 + kh * 8 + (si - 8));
          hwr[(size_t)batch * KPAD_H + unit] = f2bf(hv);
          if (phase) decH[(size_t)s * BB * HH + batch * HH + unit] = hv;
        }
      }

      // grid-wide barrier (all 125 single-wave blocks resident)
      __threadfence();
      if (lane == 0) {
        atomicAdd(&bar[gs], 1u);
        while (atomicAdd(&bar[gs], 0u) < (unsigned)gridDim.x) { __builtin_amdgcn_s_sleep(1); }
      }
      __threadfence();
    }
  }
}

// ---------------- fused projection + online log-softmax + NLL ----------------
// One block per timestep t (256 blocks): stages the 32 h-rows in LDS and amortizes
// each out_W row over 16 batches at a time (2 passes) -> out_W L2 traffic cut 16x.
// All lanes read identical LDS addresses in the inner loop => pure broadcast.
__global__ void __launch_bounds__(256) k_loss(const float* __restrict__ seq,
                                              const int* __restrict__ tgt,
                                              const float* __restrict__ decH,
                                              const float* __restrict__ outW,
                                              const float* __restrict__ outB,
                                              float* __restrict__ out) {
  __shared__ __align__(16) float hs[BB][HH];   // 64000 B
  __shared__ int   st[BB];
  __shared__ float stl[16];
  __shared__ float rmx[8], rsm[8];
  int t = blockIdx.x;
  int tid = threadIdx.x;
  int lane = tid & 31, wid = tid >> 5;

  if (t > 0) {
    for (int i = tid; i < BB * HH; i += 256)
      ((float*)hs)[i] = decH[(size_t)(t - 1) * BB * HH + i];
  }
  if (tid < BB) st[tid] = tgt[t * BB + tid];
  __syncthreads();

  float nllsum = 0.0f;                         // meaningful on tid 0 only
  for (int bg = 0; bg < BB; bg += 16) {        // two batch groups of 16
    if (tid < 16) stl[tid] = 0.0f;
    __syncthreads();

    float mx[16], sm[16];
#pragma unroll
    for (int b = 0; b < 16; ++b) { mx[b] = -3.0e38f; sm[b] = 0.0f; }

    for (int col = tid; col < VV; col += 256) {
      float lg[16];
      if (t == 0) {
#pragma unroll
        for (int b = 0; b < 16; ++b) lg[b] = seq[(size_t)(bg + b) * VV + col];
      } else {
#pragma unroll
        for (int b = 0; b < 16; ++b) lg[b] = 0.0f;
        const float4* w = (const float4*)(outW + (size_t)col * HH);
        for (int i = 0; i < 125; ++i) {
          float4 wv = w[i];
#pragma unroll
          for (int b = 0; b < 16; ++b) {
            float4 hv = *(const float4*)&hs[bg + b][i * 4];   // LDS broadcast
            lg[b] += wv.x * hv.x + wv.y * hv.y + wv.z * hv.z + wv.w * hv.w;
          }
        }
        float bb = outB[col];
#pragma unroll
        for (int b = 0; b < 16; ++b) lg[b] += bb;
      }
#pragma unroll
      for (int b = 0; b < 16; ++b) {
        float logit = lg[b];
        if (col == st[bg + b]) stl[b] = logit;
        if (logit > mx[b]) { sm[b] = sm[b] * __expf(mx[b] - logit) + 1.0f; mx[b] = logit; }
        else               { sm[b] += __expf(logit - mx[b]); }
      }
    }

    // per-batch block reduction: wave-level shuffle merge, then 8 wave results via LDS
    for (int b = 0; b < 16; ++b) {
      float m = mx[b], s = sm[b];
#pragma unroll
      for (int off = 16; off > 0; off >>= 1) {
        float om = __shfl_xor(m, off, 32);
        float os = __shfl_xor(s, off, 32);
        float M = fmaxf(m, om);
        s = s * __expf(m - M) + os * __expf(om - M);
        m = M;
      }
      if (lane == 0) { rmx[wid] = m; rsm[wid] = s; }
      __syncthreads();
      if (tid == 0) {
        float M = -3.0e38f;
        for (int i = 0; i < 8; ++i) M = fmaxf(M, rmx[i]);
        float S = 0.0f;
        for (int i = 0; i < 8; ++i) S += rsm[i] * __expf(rmx[i] - M);
        nllsum += -(stl[b] - M - logf(S));
      }
      __syncthreads();
    }
  }
  if (tid == 0) atomicAdd(out, nllsum * (1.0f / (TT * BB)));
}

// ---------------- host side ----------------
extern "C" void kernel_launch(void* const* d_in, const int* in_sizes, int n_in,
                              void* d_out, int out_size, void* d_ws, size_t ws_size,
                              hipStream_t stream) {
  (void)in_sizes; (void)n_in; (void)out_size; (void)ws_size;
  const float* seq    = (const float*)d_in[0];
  const int*   nseq   = (const int*)  d_in[1];
  const float* encWih = (const float*)d_in[2];
  const float* encWhh = (const float*)d_in[3];
  const float* encBih = (const float*)d_in[4];
  const float* encBhh = (const float*)d_in[5];
  const float* decWih = (const float*)d_in[6];
  const float* decWhh = (const float*)d_in[7];
  const float* decBih = (const float*)d_in[8];
  const float* decBhh = (const float*)d_in[9];
  const float* outW   = (const float*)d_in[10];
  const float* outB   = (const float*)d_in[11];
  float* out = (float*)d_out;

  // workspace carve-out (~270 MB total)
  char* ws = (char*)d_ws;
  size_t off = 0;
  auto alloc = [&](size_t bytes) -> char* {
    char* p = ws + off;
    off += (bytes + 255) & ~(size_t)255;
    return p;
  };
  unsigned short* seqBf   = (unsigned short*)alloc((size_t)NROWS * KPAD_V * 2);
  unsigned short* encWihp = (unsigned short*)alloc((size_t)GP * KPAD_V * 2);
  unsigned short* decWihp = (unsigned short*)alloc((size_t)GP * KPAD_V * 2);
  unsigned short* encWhhp = (unsigned short*)alloc((size_t)GG * KPAD_H * 2);
  unsigned short* decWhhp = (unsigned short*)alloc((size_t)GG * KPAD_H * 2);
  float*          encBp   = (float*)alloc((size_t)GG * 4);
  float*          decBp   = (float*)alloc((size_t)GG * 4);
  float*          gEnc    = (float*)alloc((size_t)NROWS * GP * 4);
  float*          gDec    = (float*)alloc((size_t)NROWS * GP * 4);
  unsigned short* hbuf    = (unsigned short*)alloc((size_t)2 * BB * KPAD_H * 2);
  float*          decH    = (float*)alloc((size_t)STEPS * BB * HH * 4);
  unsigned int*   bar     = (unsigned int*)alloc(1024 * 4);

  k_init<<<128, 256, 0, stream>>>(hbuf, bar, out);
  k_conv_seq<<<NROWS, 256, 0, stream>>>(seq, seqBf);
  k_conv_wih<<<GP, 256, 0, stream>>>(encWih, encWihp);
  k_conv_wih<<<GP, 256, 0, stream>>>(decWih, decWihp);
  k_conv_whh<<<GG, 256, 0, stream>>>(encWhh, encWhhp);
  k_conv_whh<<<GG, 256, 0, stream>>>(decWhh, decWhhp);
  k_bias<<<(GG + 255) / 256, 256, 0, stream>>>(encBih, encBhh, encBp);
  k_bias<<<(GG + 255) / 256, 256, 0, stream>>>(decBih, decBhh, decBp);

  dim3 g2(GP / 256, NROWS / 64);   // (8, 128)
  wmma_xgates<<<g2, 256, 0, stream>>>(seqBf, encWihp, gEnc);
  wmma_xgates<<<g2, 256, 0, stream>>>(seqBf, decWihp, gDec);

  lstm_recurrence<<<NTILES_G, 32, 0, stream>>>(encWhhp, decWhhp, encBp, decBp,
                                               gEnc, gDec, hbuf, decH, bar);

  k_loss<<<TT, 256, 0, stream>>>(seq, nseq, decH, outW, outB, out);
}
